// VanillaRNN_45878840656310
// MI455X (gfx1250) — matrix operand order repaired and checked by reading, not compiled, not run on hardware
//
#include <hip/hip_runtime.h>
#include <hip/hip_bf16.h>
#include <math.h>

// ---------------------------------------------------------------------------
// VanillaRNN on MI455X (gfx1250):
//   - persistent kernel, 16 WGs x 256 threads (8 wave32 per WG)
//   - W_hh^T (bf16) slice cached in LDS once (132 KB/WGP, LDS is 320 KB),
//     staged with GLOBAL_LOAD_ASYNC_TO_LDS_B128 (ASYNCcnt-tracked)
//   - per step: 128x1024x1024 GEMM via v_wmma_f32_16x16x32_bf16, f32 accum
//   - ping-pong bf16 h buffers in L2; monotonic atomic barrier per step
// ---------------------------------------------------------------------------

typedef __attribute__((ext_vector_type(16))) __bf16 v16bf;
typedef __attribute__((ext_vector_type(8)))  __bf16 v8bf;
typedef __attribute__((ext_vector_type(8)))  float  v8f;
typedef int v4i_vs __attribute__((vector_size(16)));   // matches builtin param

#define GLOBAL_AS __attribute__((address_space(1)))
#define LDS_AS    __attribute__((address_space(3)))

static constexpr int BATCH = 128;
static constexpr int SEQ   = 1024;
static constexpr int HID   = 1024;
static constexpr int NCLS  = 10;
static constexpr int NWG   = 16;        // persistent workgroups
static constexpr int TPB   = 256;       // 8 waves of 32
static constexpr int NCOLS = HID / NWG; // 64 hidden columns per WG
static constexpr int LDS_ROW = HID + 8; // 1032 bf16 per row -> 2064B = 516 dw,
                                        // lane advance == 4 banks: conflict-free b128
static constexpr int LDS_BYTES = NCOLS * LDS_ROW * 2; // 132096 B

// ---- one-time: W_hh (f32, [k][n]) -> Wt (bf16, [n][k]) ---------------------
__global__ void wt_transpose_kernel(const float* __restrict__ W, __bf16* __restrict__ Wt) {
    int idx = blockIdx.x * TPB + threadIdx.x;     // 1M threads
    int n = idx >> 10;
    int k = idx & (HID - 1);
    Wt[(size_t)n * HID + k] = (__bf16)W[(size_t)k * HID + n];
}

// ---- per-call: zero h0 (bf16) and the global barrier counter ---------------
__global__ void init_state_kernel(unsigned* __restrict__ h0u, unsigned* __restrict__ ctr) {
    int idx = blockIdx.x * TPB + threadIdx.x;
    if (idx < BATCH * HID / 2) h0u[idx] = 0u;     // 65536 dwords of bf16 zeros
    if (idx == 0) *ctr = 0u;
}

// ---- persistent RNN kernel --------------------------------------------------
__global__ void __launch_bounds__(TPB, 1)
rnn_persistent_kernel(const float* __restrict__ x,     // [128,1024]
                      const float* __restrict__ W_hx,  // [1,1024]
                      const float* __restrict__ B_h,   // [1024]
                      const float* __restrict__ W_ph,  // [1024,10]
                      const float* __restrict__ B_y,   // [10]
                      const __bf16* __restrict__ Wt,   // [n][k] bf16
                      __bf16* __restrict__ hbuf0,      // [128,1024] bf16
                      __bf16* __restrict__ hbuf1,
                      unsigned* __restrict__ ctr,
                      float* __restrict__ out)         // [128,10] f32
{
    extern __shared__ char smem[];
    __bf16* ldsW = (__bf16*)smem;

    const int tid    = threadIdx.x;
    const int wg     = blockIdx.x;          // 0..15
    const int wgBase = wg * NCOLS;          // first hidden column owned
    const int lane   = tid & 31;
    const int wave   = tid >> 5;            // 0..7 -> M tile
    const int laneN  = lane & 15;
    const int hiHalf = lane >> 4;           // 0: lanes 0-15, 1: lanes 16-31
    const int mBase  = wave * 16;

    // --- stage this WG's W_hh^T slice into LDS once (step-invariant B) ------
#if __has_builtin(__builtin_amdgcn_global_load_async_to_lds_b128) && \
    __has_builtin(__builtin_amdgcn_s_wait_asynccnt)
    {
        // 64 rows x 2048 B each, LDS rows padded to 2064 B; 16 B per async op.
        const char* gbase = (const char*)(Wt + (size_t)wgBase * HID);
        for (int p = tid; p < NCOLS * 128; p += TPB) {     // 8192 16B chunks
            int nl = p >> 7;                               // row 0..63
            int c16 = p & 127;                             // 16B chunk in row
            GLOBAL_AS v4i_vs* g =
                (GLOBAL_AS v4i_vs*)(gbase + (size_t)nl * 2048 + c16 * 16);
            LDS_AS v4i_vs* l =
                (LDS_AS v4i_vs*)((LDS_AS char*)(LDS_AS void*)smem + nl * 2064 + c16 * 16);
            __builtin_amdgcn_global_load_async_to_lds_b128(g, l, 0, 0);
        }
        __builtin_amdgcn_s_wait_asynccnt(0);
    }
#else
    {
        const unsigned* gsrc = (const unsigned*)(Wt + (size_t)wgBase * HID);
        unsigned* ldsu = (unsigned*)smem;
        #pragma unroll 4
        for (int p = tid; p < NCOLS * (HID / 2); p += TPB) {
            int nl = p >> 9;                // row (0..63), 512 dwords per row
            int kp = p & 511;
            ldsu[nl * (LDS_ROW / 2) + kp] = gsrc[nl * (HID / 2) + kp];
        }
    }
#endif
    __syncthreads();

    // --- per-lane, step-invariant epilogue constants ------------------------
    float whxc[4], bhc[4];
    int   colG[4];
    #pragma unroll
    for (int t = 0; t < 4; ++t) {
        colG[t] = wgBase + t * 16 + laneN;
        whxc[t] = W_hx[colG[t]];
        bhc[t]  = B_h[colG[t]];
    }
    const int rowBase = mBase + (hiHalf ? 8 : 0);   // C-layout rows: rowBase+j
    const int aRow    = mBase + laneN;              // A-layout row for this lane

    // Two LDS base pointers so all four N-tiles are reachable with 16-bit
    // immediate DS offsets (0 and 16*LDS_ROW*2 = 33024 B from each base).
    const __bf16* bpA = ldsW + laneN * LDS_ROW + (hiHalf ? 16 : 0);  // tiles 0,1
    const __bf16* bpB = bpA + 32 * LDS_ROW;                          // tiles 2,3
    constexpr int TILE_STRIDE = 16 * LDS_ROW;                        // elements

    const v8f vzero = {0.f, 0.f, 0.f, 0.f, 0.f, 0.f, 0.f, 0.f};

    // --- time-step loop -----------------------------------------------------
    for (int s = 0; s < SEQ; ++s) {
        const __bf16* __restrict__ hp = (s & 1) ? hbuf1 : hbuf0;
        __bf16* __restrict__       hn = (s & 1) ? hbuf0 : hbuf1;

        v8f acc[4];
        #pragma unroll
        for (int t = 0; t < 4; ++t) acc[t] = vzero;

        // A: 16x32 bf16 tile of h (k = 16*half + s, same order as B); B: four 32x16 tiles in LDS
        const __bf16* aBase = hp + (size_t)aRow * HID + (hiHalf ? 16 : 0);
        #pragma unroll 4
        for (int k0 = 0; k0 < HID; k0 += 32) {
            v8bf alo = *(const v8bf*)(aBase + k0);        // K k0..k0+7   (+16 hi)
            v8bf ahi = *(const v8bf*)(aBase + k0 + 8);    // K k0+8..+15  (+16 hi)
            v16bf a = __builtin_shufflevector(alo, ahi,
                        0,1,2,3,4,5,6,7,8,9,10,11,12,13,14,15);

            const __bf16* p0 = bpA + k0;                  // tiles 0,1 window
            const __bf16* p2 = bpB + k0;                  // tiles 2,3 window
            #pragma unroll
            for (int t = 0; t < 4; ++t) {
                const __bf16* bp = ((t & 2) ? p2 : p0) + ((t & 1) ? TILE_STRIDE : 0);
                v8bf blo = *(const v8bf*)(bp);            // 16 contiguous K
                v8bf bhi = *(const v8bf*)(bp + 8);
                v16bf b = __builtin_shufflevector(blo, bhi,
                            0,1,2,3,4,5,6,7,8,9,10,11,12,13,14,15);
                acc[t] = __builtin_amdgcn_wmma_f32_16x16x32_bf16(
                            false, a, false, b, (short)0, acc[t], false, false);
            }
        }

        // epilogue: + x_t*W_hx + B_h, tanh, write bf16 h_next
        float xr[8];
        #pragma unroll
        for (int j = 0; j < 8; ++j)
            xr[j] = x[(size_t)(rowBase + j) * SEQ + s];
        #pragma unroll
        for (int t = 0; t < 4; ++t) {
            #pragma unroll
            for (int j = 0; j < 8; ++j) {
                float v = acc[t][j] + xr[j] * whxc[t] + bhc[t];
                hn[(size_t)(rowBase + j) * HID + colG[t]] = (__bf16)tanhf(v);
            }
        }

        // device-wide step barrier: monotonic counter (no ABA, no reset)
        __threadfence();
        __syncthreads();
        if (tid == 0)
            __hip_atomic_fetch_add(ctr, 1u, __ATOMIC_ACQ_REL, __HIP_MEMORY_SCOPE_AGENT);
        const unsigned target = (unsigned)(NWG * (s + 1));
        while (__hip_atomic_load(ctr, __ATOMIC_ACQUIRE, __HIP_MEMORY_SCOPE_AGENT) < target)
            __builtin_amdgcn_s_sleep(1);
    }

    // --- classifier + softmax on WG0 (h_T is in hbuf0 since SEQ is even) ----
    if (wg != 0) return;
    const __bf16* hT = hbuf0;
    float* lsLog = (float*)smem;            // reuse LDS (weights no longer read)
    __syncthreads();
    for (int p = tid; p < BATCH * NCLS; p += TPB) {
        int b = p / NCLS, c = p % NCLS;
        const __bf16* hr = hT + (size_t)b * HID;
        float sum = B_y[c];
        for (int k = 0; k < HID; ++k)
            sum += (float)hr[k] * W_ph[(size_t)k * NCLS + c];
        lsLog[p] = sum;
    }
    __syncthreads();
    if (tid < BATCH) {
        float m = lsLog[tid * NCLS];
        #pragma unroll
        for (int c = 1; c < NCLS; ++c) m = fmaxf(m, lsLog[tid * NCLS + c]);
        float e[NCLS], sum = 0.f;
        #pragma unroll
        for (int c = 0; c < NCLS; ++c) { e[c] = __expf(lsLog[tid * NCLS + c] - m); sum += e[c]; }
        float inv = 1.0f / sum;
        #pragma unroll
        for (int c = 0; c < NCLS; ++c) out[tid * NCLS + c] = e[c] * inv;
    }
}

// ---------------------------------------------------------------------------
extern "C" void kernel_launch(void* const* d_in, const int* in_sizes, int n_in,
                              void* d_out, int out_size, void* d_ws, size_t ws_size,
                              hipStream_t stream) {
    const float* x    = (const float*)d_in[0];   // [128,1024]
    const float* W_hx = (const float*)d_in[1];   // [1,1024]
    const float* W_hh = (const float*)d_in[2];   // [1024,1024]
    const float* B_h  = (const float*)d_in[3];   // [1024]
    const float* W_ph = (const float*)d_in[4];   // [1024,10]
    const float* B_y  = (const float*)d_in[5];   // [10]
    float* out = (float*)d_out;

    // workspace layout
    char* ws = (char*)d_ws;
    __bf16* Wt    = (__bf16*)(ws);                                  // 2 MB
    __bf16* hbuf0 = (__bf16*)(ws + (size_t)HID * HID * 2);          // 256 KB
    __bf16* hbuf1 = (__bf16*)(ws + (size_t)HID * HID * 2 + (size_t)BATCH * HID * 2);
    unsigned* ctr = (unsigned*)(ws + (size_t)HID * HID * 2 + 2 * (size_t)BATCH * HID * 2);

    // allow 132 KB dynamic LDS (WGP has 320 KB)
    (void)hipFuncSetAttribute((const void*)rnn_persistent_kernel,
                              hipFuncAttributeMaxDynamicSharedMemorySize, LDS_BYTES);

    wt_transpose_kernel<<<(HID * HID) / TPB, TPB, 0, stream>>>(W_hh, Wt);
    init_state_kernel<<<(BATCH * HID / 2 + TPB - 1) / TPB, TPB, 0, stream>>>(
        (unsigned*)hbuf0, ctr);
    rnn_persistent_kernel<<<NWG, TPB, LDS_BYTES, stream>>>(
        x, W_hx, B_h, W_ph, B_y, Wt, hbuf0, hbuf1, ctr, out);
}
